// Structure_Module_47390669144111
// MI455X (gfx1250) — matrix operand order repaired and checked by reading, not compiled, not run on hardware
//
#include <hip/hip_runtime.h>
#include <hip/hip_bf16.h>

// ---------------------------------------------------------------------------
// Structure module for gfx1250 (MI455X, wave32, WMMA).
//   - z is dead in the reference -> never touched (saves ~300MB of traffic).
//   - All 384x384 GEMMs run on v_wmma_f32_16x16x32_f16 (f16 A/B, f32 acc).
//   - A-stripe staged per-block in LDS via GLOBAL_LOAD_ASYNC_TO_LDS_B128
//     (ASYNCcnt path), converted f32->f16 once cooperatively, fragments
//     fetched with ds_load_b128 from a bank-padded tile.
//   - Weights pre-converted once to transposed f16 [N][K] in workspace.
//   - Working set (~14MB) is fully L2-resident; kernels are compute bound.
// ---------------------------------------------------------------------------

#define B_  4
#define R_  384
#define CS  384
#define M_  (B_*R_)      // 1536 rows
#define NL  8

typedef __attribute__((ext_vector_type(16))) _Float16 v16h;
typedef __attribute__((ext_vector_type(8)))  _Float16 h8;
typedef __attribute__((ext_vector_type(2)))  _Float16 h2;
typedef __attribute__((ext_vector_type(8)))  float    v8f;
typedef __attribute__((ext_vector_type(4)))  float    f32x4;

union V16H { v16h v; _Float16 h[16]; h8 q[2]; f32x4 f4[2]; };

// ---------------------------------------------------------------------------
// GEMM: Out[M,N] = op(A[M,K]) @ Wt^T + bias (+ Res) (opt relu)
// M=1536, N=K=384.  Wt is f16, layout [N][K] (pre-transposed).
// Block = 8 waves = one 16-row stripe x 8 column tiles (128 cols).
// 288 blocks = 96 stripe-rows x 3 column groups.
// ---------------------------------------------------------------------------
template<bool RELU_IN, bool RELU_OUT, bool RESID>
__global__ __launch_bounds__(256) void gemm384(const float* __restrict__ A,
                                               const _Float16* __restrict__ Wt,
                                               const float* __restrict__ bias,
                                               const float* __restrict__ Res,
                                               float* __restrict__ Out)
{
  constexpr int K = 384, N = 384;
  constexpr int LDH = 392;                 // padded f16 row stride (halves):
                                           // 392*2B = 784B = 196 dwords; 196%64=4
  __shared__ float    stageF[16 * K];      // 24576 B : async-load landing zone
  __shared__ _Float16 tileH[16 * LDH];     // 12544 B : shared f16 A stripe

  const int tid  = threadIdx.x;
  const int wave = tid >> 5;
  const int lane = tid & 31;
  const int tileRow = blockIdx.x / 3;                 // 0..95
  const int colTile = (blockIdx.x % 3) * 8 + wave;    // 0..23
  const int m0 = tileRow * 16;
  const int n0 = colTile * 16;
  const int kh = lane >> 4;                // lane half (0/1)
  const int nc = n0 + (lane & 15);         // output column this lane holds

  // ---- Phase 1: async global->LDS copy of the 16xK f32 A stripe ----------
  // 24576 B / 256 threads = 6 x b128 per thread.  GVS mode: saddr = uniform
  // stripe base, vaddr = 32-bit byte offset, vdst = LDS byte address.
  {
    const unsigned long long gbase = (unsigned long long)(A + (size_t)m0 * K);
    const unsigned ldsF = (unsigned)(size_t)(&stageF[0]);
    #pragma unroll
    for (int i = 0; i < 6; ++i) {
      const unsigned boff = (unsigned)(tid * 16 + i * 4096);
      const unsigned dst  = ldsF + boff;
      asm volatile("global_load_async_to_lds_b128 %0, %1, %2"
                   :: "v"(dst), "v"(boff), "s"(gbase) : "memory");
    }
    asm volatile("s_wait_asynccnt 0" ::: "memory");
  }
  __syncthreads();

  // ---- Phase 2: cooperative f32 -> f16 convert once (folds RELU_IN) ------
  #pragma unroll
  for (int i = 0; i < 12; ++i) {
    const int v2 = (tid + i * 256) * 2;       // 6144 values total
    const int m = v2 / K, k = v2 % K;
    float f0 = stageF[v2], f1 = stageF[v2 + 1];
    if (RELU_IN) { f0 = fmaxf(f0, 0.f); f1 = fmaxf(f1, 0.f); }
    h2 p; p.x = (_Float16)f0; p.y = (_Float16)f1;
    *(h2*)(&tileH[m * LDH + k]) = p;          // 4B aligned (784%4==0, k even)
  }
  __syncthreads();

  // ---- Phase 3: WMMA main loop -------------------------------------------
  // A fragment (contiguous k order, matching B):
  //   element e : K = kk + kh*16 + e
  // -> two contiguous 8-half runs = two ds_load_b128 (16B aligned).
  const _Float16* hrow = &tileH[(lane & 15) * LDH];
  const _Float16* brow = Wt + (size_t)nc * K + kh * 16;
  v8f acc = {};
  #pragma unroll 4
  for (int kk = 0; kk < K; kk += 32) {
    V16H af;
    af.q[0] = *(const h8*)(hrow + kk + kh * 16);
    af.q[1] = *(const h8*)(hrow + kk + kh * 16 + 8);
    // B fragment: lane half kh holds K = kk + kh*16 + 0..15, contiguous in Wt.
    V16H bf;
    const f32x4* bp = (const f32x4*)(brow + kk);
    bf.f4[0] = bp[0]; bf.f4[1] = bp[1];
    acc = __builtin_amdgcn_wmma_f32_16x16x32_f16(false, af.v, false, bf.v,
                                                 (short)0, acc, false, false);
  }

  // C/D layout: vgpr v -> row m0 + v + 8*kh, col = n0 + (lane&15)
  const float bn = bias[nc];
  #pragma unroll
  for (int vv = 0; vv < 8; ++vv) {
    const int m = m0 + vv + 8 * kh;
    const size_t off = (size_t)m * N + nc;
    float r = acc[vv] + bn;
    if (RESID)    r += Res[off];
    if (RELU_OUT) r = fmaxf(r, 0.f);
    Out[off] = r;
  }
}

// ---------------------------------------------------------------------------
// Row layernorm over 384; one block (128 thr) per row; safe in-place.
// ---------------------------------------------------------------------------
__global__ __launch_bounds__(128) void layernorm384(const float* __restrict__ X,
                                                    const float* __restrict__ g,
                                                    const float* __restrict__ bt,
                                                    float* __restrict__ Y)
{
  __shared__ float red[128];
  const int row = blockIdx.x, t = threadIdx.x;
  const float* x = X + (size_t)row * CS;
  float v0 = x[t], v1 = x[t + 128], v2 = x[t + 256];
  red[t] = v0 + v1 + v2; __syncthreads();
  for (int s2 = 64; s2 > 0; s2 >>= 1) { if (t < s2) red[t] += red[t + s2]; __syncthreads(); }
  const float mean = red[0] * (1.f / CS);
  __syncthreads();
  red[t] = (v0-mean)*(v0-mean) + (v1-mean)*(v1-mean) + (v2-mean)*(v2-mean);
  __syncthreads();
  for (int s2 = 64; s2 > 0; s2 >>= 1) { if (t < s2) red[t] += red[t + s2]; __syncthreads(); }
  const float inv = rsqrtf(red[0] * (1.f / CS) + 1e-5f);
  float* y = Y + (size_t)row * CS;
  y[t]       = (v0 - mean) * inv * g[t]       + bt[t];
  y[t + 128] = (v1 - mean) * inv * g[t + 128] + bt[t + 128];
  y[t + 256] = (v2 - mean) * inv * g[t + 256] + bt[t + 256];
}

// ---------------------------------------------------------------------------
// Init: bb_r = I, bb_t = 0, loss accumulators = 0 (ws is poisoned by harness).
// ---------------------------------------------------------------------------
__global__ void init_state(float* bbr, float* bbt, float* sums)
{
  const int i = blockIdx.x * blockDim.x + threadIdx.x;
  if (i < 3 * NL) sums[i] = 0.f;
  if (i < M_) {
    float* Rp = bbr + (size_t)i * 9;
    Rp[0]=1.f; Rp[1]=0.f; Rp[2]=0.f;
    Rp[3]=0.f; Rp[4]=1.f; Rp[5]=0.f;
    Rp[6]=0.f; Rp[7]=0.f; Rp[8]=1.f;
    bbt[i*3+0]=0.f; bbt[i*3+1]=0.f; bbt[i*3+2]=0.f;
  }
}

// Convert+transpose 7 weight matrices f32 [K][N] -> f16 [N][K].
struct WSrc { const float* p[7]; };
__global__ void prep_weights(WSrc w, _Float16* Wt)
{
  const int gid = blockIdx.x * blockDim.x + threadIdx.x;   // exactly 7*384*384
  const int widx = gid / (CS * CS);
  const int r    = gid % (CS * CS);
  const int n = r / CS, k = r % CS;
  Wt[(size_t)widx * CS * CS + (size_t)n * CS + k] = (_Float16)w.p[widx][(size_t)k * CS + n];
}

// ---------------------------------------------------------------------------
// Backbone update: q = s @ bb_w + bb_b ; quat->rot ; compose.
// ---------------------------------------------------------------------------
__global__ __launch_bounds__(256) void backbone_update(const float* __restrict__ s,
                                                       const float* __restrict__ bbw,
                                                       const float* __restrict__ bbb,
                                                       float* __restrict__ bbr,
                                                       float* __restrict__ bbt)
{
  const int i = blockIdx.x * blockDim.x + threadIdx.x;   // exactly 1536
  const float* srow = s + (size_t)i * CS;
  float q[6];
  #pragma unroll
  for (int j = 0; j < 6; ++j) q[j] = bbb[j];
  for (int k = 0; k < CS; ++k) {
    const float sv = srow[k];
    const float* wr = bbw + k * 6;
    #pragma unroll
    for (int j = 0; j < 6; ++j) q[j] += sv * wr[j];
  }
  float b = q[0], c = q[1], d = q[2];
  const float n = sqrtf(1.f + b*b + c*c + d*d);
  const float a = 1.f / n; b /= n; c /= n; d /= n;
  float nr[9];
  nr[0]=a*a+b*b-c*c-d*d; nr[1]=2.f*(b*c-a*d);     nr[2]=2.f*(b*d+a*c);
  nr[3]=2.f*(b*c+a*d);   nr[4]=a*a-b*b+c*c-d*d;   nr[5]=2.f*(c*d-a*b);
  nr[6]=2.f*(b*d-a*c);   nr[7]=2.f*(c*d+a*b);     nr[8]=a*a-b*b-c*c+d*d;
  float* Rp = bbr + (size_t)i * 9;
  float Ro[9];
  #pragma unroll
  for (int x = 0; x < 9; ++x) Ro[x] = Rp[x];
  #pragma unroll
  for (int x = 0; x < 3; ++x)
    #pragma unroll
    for (int z = 0; z < 3; ++z)
      Rp[x*3+z] = Ro[x*3+0]*nr[z] + Ro[x*3+1]*nr[3+z] + Ro[x*3+2]*nr[6+z];
  bbt[i*3+0] += q[3]; bbt[i*3+1] += q[4]; bbt[i*3+2] += q[5];
}

// ---------------------------------------------------------------------------
// Torsion head: a7 (384x4) on relu(a); a_norm out; accumulate L_tor/L_ang sums.
// ---------------------------------------------------------------------------
__global__ __launch_bounds__(256) void torsion(const float* __restrict__ a,
                                               const float* __restrict__ w7,
                                               const float* __restrict__ b7,
                                               const float* __restrict__ labels,
                                               float* __restrict__ a_norm_out,
                                               float* __restrict__ tor_sum,
                                               float* __restrict__ ang_sum)
{
  __shared__ float red[256];
  const int i = blockIdx.x * blockDim.x + threadIdx.x;   // exactly 1536
  const int t = threadIdx.x;
  const float* arow = a + (size_t)i * CS;
  float o[4] = { b7[0], b7[1], b7[2], b7[3] };
  for (int k = 0; k < CS; ++k) {
    const float av = fmaxf(arow[k], 0.f);
    const float* wr = w7 + k * 4;
    o[0] += av*wr[0]; o[1] += av*wr[1]; o[2] += av*wr[2]; o[3] += av*wr[3];
  }
  const float lphi = sqrtf(o[0]*o[0] + o[1]*o[1]);
  const float lpsi = sqrtf(o[2]*o[2] + o[3]*o[3]);
  const float an[4] = { o[0]/lphi, o[1]/lphi, o[2]/lpsi, o[3]/lpsi };
  float tor = 0.f;
  #pragma unroll
  for (int j = 0; j < 4; ++j) {
    a_norm_out[(size_t)i*4 + j] = an[j];
    const float dl = an[j] - labels[(size_t)i*4 + j];
    tor += dl * dl;
  }
  const float ang = fabsf(lphi - 1.f) + fabsf(lpsi - 1.f);
  red[t] = tor; __syncthreads();
  for (int s2 = 128; s2 > 0; s2 >>= 1) { if (t < s2) red[t] += red[t + s2]; __syncthreads(); }
  if (!t) atomicAdd(tor_sum, red[0]);
  __syncthreads();
  red[t] = ang; __syncthreads();
  for (int s2 = 128; s2 > 0; s2 >>= 1) { if (t < s2) red[t] += red[t + s2]; __syncthreads(); }
  if (!t) atomicAdd(ang_sum, red[0]);
}

// ---------------------------------------------------------------------------
// FAPE: sum over (b,i,j,xyz) of min(sqrt((Ri tj + ti - (Rli tj + tli))^2+eps),10)
// ---------------------------------------------------------------------------
__global__ __launch_bounds__(256) void fape(const float* __restrict__ bbr,
                                            const float* __restrict__ bbt,
                                            const float* __restrict__ Rl,
                                            const float* __restrict__ tl,
                                            float* __restrict__ fape_sum)
{
  __shared__ float red[256];
  const int gid = blockIdx.x * blockDim.x + threadIdx.x;   // exactly B*R*R
  const int b = gid / (R_ * R_);
  const int rem = gid % (R_ * R_);
  const int i = rem / R_, j = rem % R_;
  const int bi = b * R_ + i, bj = b * R_ + j;
  const float* Ri  = bbr + (size_t)bi * 9;
  const float* ti  = bbt + (size_t)bi * 3;
  const float* tj  = bbt + (size_t)bj * 3;   // x == current bb_t in reference
  const float* Rli = Rl  + (size_t)bi * 9;
  const float* tli = tl  + (size_t)bi * 3;
  const float p0 = tj[0], p1 = tj[1], p2 = tj[2];
  float s = 0.f;
  #pragma unroll
  for (int x = 0; x < 3; ++x) {
    const float xc = Ri [x*3+0]*p0 + Ri [x*3+1]*p1 + Ri [x*3+2]*p2 + ti [x];
    const float xl = Rli[x*3+0]*p0 + Rli[x*3+1]*p1 + Rli[x*3+2]*p2 + tli[x];
    const float df = xc - xl;
    s += fminf(sqrtf(df * df + 1e-12f), 10.f);
  }
  const int t = threadIdx.x;
  red[t] = s; __syncthreads();
  for (int s2 = 128; s2 > 0; s2 >>= 1) { if (t < s2) red[t] += red[t + s2]; __syncthreads(); }
  if (!t) atomicAdd(fape_sum, red[0]);
}

// ---------------------------------------------------------------------------
// Finalize: pack s / bb_r / bb_t into d_out; compute mean(l_aux).
// ---------------------------------------------------------------------------
__global__ void finalize_copy(const float* __restrict__ s, const float* __restrict__ bbr,
                              const float* __restrict__ bbt, float* __restrict__ out)
{
  const int gid = blockIdx.x * blockDim.x + threadIdx.x;
  const int NS = M_ * CS, NR = M_ * 9, NT = M_ * 3;
  if      (gid < NS)           out[gid] = s[gid];
  else if (gid < NS + NR)      out[gid] = bbr[gid - NS];
  else if (gid < NS + NR + NT) out[gid] = bbt[gid - NS - NR];
}

__global__ void finalize_loss(const float* __restrict__ sums, float* __restrict__ out)
{
  if (threadIdx.x == 0 && blockIdx.x == 0) {
    float acc = 0.f;
    for (int l = 0; l < NL; ++l) {
      const float fp = sums[l];            // fape sum over B*R*R*3
      const float tr = sums[NL + l];       // torsion sq-err sum over B*R*4
      const float an = sums[2*NL + l];     // angle-norm sum over B*R
      acc += 0.1f * fp / (float)(M_ * R_ * 3)
           + tr / (float)(M_ * 4)
           + 0.02f * an / (float)M_;
    }
    out[M_*CS + M_*9 + M_*3 + M_*4] = acc / (float)NL;
  }
}

// ---------------------------------------------------------------------------
extern "C" void kernel_launch(void* const* d_in, const int* in_sizes, int n_in,
                              void* d_out, int out_size, void* d_ws, size_t ws_size,
                              hipStream_t stream)
{
  (void)in_sizes; (void)n_in; (void)out_size; (void)ws_size;
  // d_in[0] is z -> dead in the reference graph, intentionally never read.
  const float* s_i        = (const float*)d_in[1];
  const float* a_labels   = (const float*)d_in[2];
  const float* bb_r_lab   = (const float*)d_in[3];
  const float* bb_t_lab   = (const float*)d_in[4];
  // params in setup_inputs() dict insertion order:
  const float* ln_si_g = (const float*)d_in[5];
  const float* ln_si_b = (const float*)d_in[6];
  // 7,8: ln_z_g / ln_z_b (dead)
  const float* ln_ipa_g = (const float*)d_in[9];
  const float* ln_ipa_b = (const float*)d_in[10];
  const float* ln_s_g   = (const float*)d_in[11];
  const float* ln_s_b   = (const float*)d_in[12];
  const float* lin_s_w  = (const float*)d_in[13];
  const float* lin_s_b  = (const float*)d_in[14];
  const float* a1_w = (const float*)d_in[15];  const float* a1_b = (const float*)d_in[16];
  const float* a2_w = (const float*)d_in[17];  const float* a2_b = (const float*)d_in[18];
  const float* a3_w = (const float*)d_in[19];  const float* a3_b = (const float*)d_in[20];
  const float* a4_w = (const float*)d_in[21];  const float* a4_b = (const float*)d_in[22];
  const float* a5_w = (const float*)d_in[23];  const float* a5_b = (const float*)d_in[24];
  const float* a6_w = (const float*)d_in[25];  const float* a6_b = (const float*)d_in[26];
  const float* a7_w = (const float*)d_in[27];  const float* a7_b = (const float*)d_in[28];
  const float* bb_w = (const float*)d_in[29];  const float* bb_b = (const float*)d_in[30];

  // Workspace carve-up (256B aligned).
  char* wsb = (char*)d_ws; size_t off = 0;
  auto carve = [&](size_t bytes) { void* p = wsb + off; off = (off + bytes + 255) & ~(size_t)255; return p; };
  float*    S    = (float*)carve((size_t)M_ * CS * 4);
  float*    SiL  = (float*)carve((size_t)M_ * CS * 4);
  float*    T1   = (float*)carve((size_t)M_ * CS * 4);
  float*    T2   = (float*)carve((size_t)M_ * CS * 4);
  float*    Aa   = (float*)carve((size_t)M_ * CS * 4);
  float*    BBR  = (float*)carve((size_t)M_ * 9 * 4);
  float*    BBT  = (float*)carve((size_t)M_ * 3 * 4);
  float*    SUMS = (float*)carve((size_t)3 * NL * 4);
  _Float16* WT   = (_Float16*)carve((size_t)7 * CS * CS * 2);

  init_state<<<6, 256, 0, stream>>>(BBR, BBT, SUMS);
  WSrc wsrc; wsrc.p[0]=lin_s_w; wsrc.p[1]=a1_w; wsrc.p[2]=a2_w; wsrc.p[3]=a3_w;
             wsrc.p[4]=a4_w;    wsrc.p[5]=a5_w; wsrc.p[6]=a6_w;
  prep_weights<<<(7 * CS * CS) / 256, 256, 0, stream>>>(wsrc, WT);

  const _Float16 *W0 = WT,            *W1 = WT + 1*CS*CS, *W2 = WT + 2*CS*CS,
                 *W3 = WT + 3*CS*CS,  *W4 = WT + 4*CS*CS, *W5 = WT + 5*CS*CS,
                 *W6 = WT + 6*CS*CS;

  layernorm384<<<M_, 128, 0, stream>>>(s_i, ln_si_g, ln_si_b, SiL);
  gemm384<false,false,false><<<288, 256, 0, stream>>>(SiL, W0, lin_s_b, nullptr, S);

  float* anorm_out = (float*)d_out + (M_*CS + M_*9 + M_*3);
  for (int l = 0; l < NL; ++l) {
    // transition block
    layernorm384<<<M_, 128, 0, stream>>>(S, ln_ipa_g, ln_ipa_b, T1);
    gemm384<false,true ,false><<<288, 256, 0, stream>>>(T1, W1, a1_b, nullptr, T2);
    gemm384<false,true ,false><<<288, 256, 0, stream>>>(T2, W2, a2_b, nullptr, T1);
    gemm384<false,false,true ><<<288, 256, 0, stream>>>(T1, W3, a3_b, S, S);
    layernorm384<<<M_, 128, 0, stream>>>(S, ln_s_g, ln_s_b, S);        // safe in-place
    // backbone
    backbone_update<<<6, 256, 0, stream>>>(S, bb_w, bb_b, BBR, BBT);
    // torsion trunk
    gemm384<false,false,false><<<288, 256, 0, stream>>>(S,   W1, a1_b, nullptr, Aa);
    gemm384<false,false,true ><<<288, 256, 0, stream>>>(SiL, W2, a2_b, Aa, Aa);
    gemm384<true ,true ,false><<<288, 256, 0, stream>>>(Aa,  W4, a4_b, nullptr, T1);
    gemm384<false,false,true ><<<288, 256, 0, stream>>>(T1,  W3, a3_b, Aa, Aa);
    gemm384<true ,true ,false><<<288, 256, 0, stream>>>(Aa,  W6, a6_b, nullptr, T1);
    gemm384<false,false,true ><<<288, 256, 0, stream>>>(T1,  W5, a5_b, Aa, Aa);
    // losses
    torsion<<<6, 256, 0, stream>>>(Aa, a7_w, a7_b, a_labels, anorm_out,
                                   SUMS + NL + l, SUMS + 2*NL + l);
    fape<<<(B_ * R_ * R_) / 256, 256, 0, stream>>>(BBR, BBT, bb_r_lab, bb_t_lab, SUMS + l);
  }

  finalize_copy<<<(M_*CS + M_*9 + M_*3 + 255) / 256, 256, 0, stream>>>(S, BBR, BBT, (float*)d_out);
  finalize_loss<<<1, 64, 0, stream>>>(SUMS, (float*)d_out);
}
